// T5BlockPipe_9620726743098
// MI455X (gfx1250) — matrix operand order repaired and checked by reading, not compiled, not run on hardware
//
#include <hip/hip_runtime.h>
#include <hip/hip_bf16.h>

// ---------------------------------------------------------------------------
// T5 encoder block for gfx1250 (MI455X): bf16 WMMA GEMMs + flash attention.
// B=4, S=1024, D=1024, H=16, DK=64, DFF=4096, NB=32, MAXD=128
// ---------------------------------------------------------------------------

typedef __attribute__((ext_vector_type(16))) __bf16 v16bf;
typedef __attribute__((ext_vector_type(8)))  float  v8f;

union Frag16 {
    unsigned int u[8];
    v16bf v;
};

__device__ __forceinline__ unsigned short f2bf(float f) {
    unsigned int u = __float_as_uint(f);
    unsigned int r = (u + 0x7FFFu + ((u >> 16) & 1u)) >> 16;
    return (unsigned short)r;
}

// ds_swizzle xor-butterfly within 32 lanes: offset = xor_mask<<10 | and_mask(0x1f)
#define SWZF(x, m) __int_as_float(__builtin_amdgcn_ds_swizzle(__float_as_int(x), (((m) << 10) | 0x1f)))

// ---------------------------------------------------------------------------
// gfx1250 async global->LDS copies (ASYNCcnt-tracked), with sync fallback
// ---------------------------------------------------------------------------
#if defined(__gfx1250__) && __has_builtin(__builtin_amdgcn_global_load_async_to_lds_b128) && \
    __has_builtin(__builtin_amdgcn_s_wait_asynccnt)
#define USE_ASYNC_LDS 1
#else
#define USE_ASYNC_LDS 0
#endif

#if USE_ASYNC_LDS
// parameter type per hipcc diagnostic: gcc-vector int4 pointers, AS1 src / AS3 dst
typedef int v4i_g __attribute__((vector_size(4 * sizeof(int))));
typedef __attribute__((address_space(1))) v4i_g* g_v4i_ptr;
typedef __attribute__((address_space(3))) v4i_g* l_v4i_ptr;
#endif

__device__ __forceinline__ void copy16_g2l(const void* g, void* l) {
#if USE_ASYNC_LDS
    // generic LDS pointer's low 32 bits are the LDS byte offset; global generic
    // pointer's value is the 64-bit global address. Cast through integers to
    // synthesize the address-space-qualified pointers the builtin expects.
    __builtin_amdgcn_global_load_async_to_lds_b128(
        (g_v4i_ptr)(unsigned long long)g,
        (l_v4i_ptr)(unsigned int)(unsigned long long)l,
        0, 0);
#else
    *(uint4*)l = *(const uint4*)g;
#endif
}

__device__ __forceinline__ void wait_g2l() {
#if USE_ASYNC_LDS
    __builtin_amdgcn_s_wait_asynccnt(0);
#endif
}

// ---------------------------------------------------------------------------
// f32 -> bf16 elementwise convert (weights)
// ---------------------------------------------------------------------------
__global__ void __launch_bounds__(256)
cvt_f32_bf16(const float* __restrict__ in, unsigned short* __restrict__ out, int n) {
    int i = blockIdx.x * 256 + threadIdx.x;
    int stride = gridDim.x * 256;
    for (; i < n; i += stride) out[i] = f2bf(in[i]);
}

// ---------------------------------------------------------------------------
// RMSNorm (T5 style: no mean subtraction) -> bf16 output. One block per row.
// ---------------------------------------------------------------------------
__global__ void __launch_bounds__(256)
rmsnorm_bf16(const float* __restrict__ x, const float* __restrict__ w,
             unsigned short* __restrict__ out, int D) {
    const int row = blockIdx.x;
    const float* xr = x + (size_t)row * D;
    float ss = 0.0f;
    for (int i = threadIdx.x; i < D; i += 256) {
        float v = xr[i];
        ss += v * v;
    }
    // wave32 butterfly reduction via ds_swizzle
    ss += SWZF(ss, 16);
    ss += SWZF(ss, 8);
    ss += SWZF(ss, 4);
    ss += SWZF(ss, 2);
    ss += SWZF(ss, 1);
    __shared__ float red[8];
    __shared__ float sscale;
    if ((threadIdx.x & 31) == 0) red[threadIdx.x >> 5] = ss;
    __syncthreads();
    if (threadIdx.x == 0) {
        float t = 0.0f;
        #pragma unroll
        for (int i = 0; i < 8; ++i) t += red[i];
        sscale = rsqrtf(t / (float)D + 1e-6f);
    }
    __syncthreads();
    const float sc = sscale;
    unsigned short* orow = out + (size_t)row * D;
    for (int i = threadIdx.x; i < D; i += 256) orow[i] = f2bf(xr[i] * sc * w[i]);
}

// ---------------------------------------------------------------------------
// Tiled bf16 GEMM:  C[M,N] = A[M,K] @ B[K,N]
//   block tile 128x128, K-step 32, 256 threads = 8 waves (2x4), each wave
//   owns 64x32 = 8 accumulators of v_wmma_f32_16x16x32_bf16.
//   A/B tiles staged via async global->LDS (gfx1250 ASYNCcnt path).
// EPI: 0 = store bf16 ; 1 = relu + store bf16 ; 2 = add f32 residual, store f32
// ---------------------------------------------------------------------------
template <int EPI>
__global__ void __launch_bounds__(256)
gemm_bf16(const unsigned short* __restrict__ A, const unsigned short* __restrict__ Bm,
          void* __restrict__ outp, const float* __restrict__ residual,
          int M, int N, int K) {
    __shared__ unsigned short lA[128][32]; // 8 KB
    __shared__ unsigned short lB[32][128]; // 8 KB

    const int tid  = threadIdx.x;
    const int wave = tid >> 5;
    const int lane = tid & 31;
    const int lm   = lane & 15;
    const int half = lane >> 4;
    const int wm   = wave >> 2;  // 0..1  (64 rows each)
    const int wn   = wave & 3;   // 0..3  (32 cols each)
    const int rowBlk = blockIdx.x * 128;
    const int colBlk = blockIdx.y * 128;

    v8f acc[4][2];
    #pragma unroll
    for (int t = 0; t < 4; ++t)
        #pragma unroll
        for (int u = 0; u < 2; ++u) acc[t][u] = (v8f){0,0,0,0,0,0,0,0};

    for (int k0 = 0; k0 < K; k0 += 32) {
        // ---- async stage of A (128x32) and B (32x128) tiles into LDS ----
        #pragma unroll
        for (int i = 0; i < 2; ++i) {
            int c  = tid + i * 256;       // 0..511
            int r  = c >> 2;              // 4 chunks of 8 bf16 per row
            int cc = (c & 3) * 8;
            copy16_g2l(A + (size_t)(rowBlk + r) * K + k0 + cc, &lA[r][cc]);
        }
        #pragma unroll
        for (int i = 0; i < 2; ++i) {
            int c  = tid + i * 256;
            int r  = c >> 4;              // 16 chunks per row
            int cc = (c & 15) * 8;
            copy16_g2l(Bm + (size_t)(k0 + r) * N + colBlk + cc, &lB[r][cc]);
        }
        // prefetch next K tile into caches (gfx1250 global_prefetch path)
        if (k0 + 32 < K) {
            __builtin_prefetch(A + (size_t)(rowBlk + (tid >> 1)) * K + k0 + 32, 0, 1);
            __builtin_prefetch(Bm + (size_t)(k0 + 32 + (tid >> 4)) * N + colBlk + (tid & 15) * 8, 0, 1);
        }
        wait_g2l();
        __syncthreads();

        // ---- B fragments (lane = N, VGPR v + half = K pair) ----
        Frag16 bfrag[2];
        #pragma unroll
        for (int u = 0; u < 2; ++u) {
            const int nb = wn * 32 + u * 16;
            #pragma unroll
            for (int v = 0; v < 8; ++v) {
                int kk = (v >> 2) * 16 + half * 8 + (v & 3) * 2;
                bfrag[u].u[v] = (unsigned int)lB[kk][nb + lm] | ((unsigned int)lB[kk + 1][nb + lm] << 16);
            }
        }
        // ---- A fragments (lane = M, VGPR v + half = K pair) ----
        #pragma unroll
        for (int t = 0; t < 4; ++t) {
            Frag16 afrag;
            const int mb = wm * 64 + t * 16 + lm;
            #pragma unroll
            for (int v = 0; v < 8; ++v) {
                int kk = (v >> 2) * 16 + half * 8 + (v & 3) * 2;
                afrag.u[v] = *(const unsigned int*)&lA[mb][kk];
            }
            #pragma unroll
            for (int u = 0; u < 2; ++u)
                acc[t][u] = __builtin_amdgcn_wmma_f32_16x16x32_bf16(
                    false, afrag.v, false, bfrag[u].v, (short)0, acc[t][u],
                    false, false);
        }
        __syncthreads();
    }

    // ---- epilogue ----
    #pragma unroll
    for (int t = 0; t < 4; ++t) {
        #pragma unroll
        for (int u = 0; u < 2; ++u) {
            #pragma unroll
            for (int r = 0; r < 8; ++r) {
                const int R  = rowBlk + wm * 64 + t * 16 + r + 8 * half;
                const int Cc = colBlk + wn * 32 + u * 16 + lm;
                float val = acc[t][u][r];
                if (EPI == 1) val = fmaxf(val, 0.0f);
                if (EPI == 2) {
                    ((float*)outp)[(size_t)R * N + Cc] =
                        val + residual[(size_t)R * N + Cc];
                } else {
                    ((unsigned short*)outp)[(size_t)R * N + Cc] = f2bf(val);
                }
            }
        }
    }
}

// ---------------------------------------------------------------------------
// T5 relative-position bucket (bidirectional, NB=32, MAXD=128)
// ---------------------------------------------------------------------------
__device__ __forceinline__ int relbucket(int q, int k) {
    int rel = k - q;
    int ret = (rel > 0) ? 16 : 0;
    int n = rel < 0 ? -rel : rel;
    if (n < 8) return ret + n;
    // 8 + log(n/8)/log(128/8)*8, clamped to 15
    int vl = 8 + (int)(__logf((float)n * 0.125f) * (8.0f / __logf(16.0f)));
    vl = vl < 15 ? vl : 15;
    return ret + vl;
}

// ---------------------------------------------------------------------------
// Flash attention: grid = (B*H, S/64). Block = 128 threads (4 waves),
// each wave owns 16 query rows. Online softmax in C-fragment layout:
// row statistics are uniform across each 16-lane half after ds_swizzle
// butterflies, so exp/rescale happen in-place with no layout change.
// ---------------------------------------------------------------------------
__global__ void __launch_bounds__(128)
flash_attn(const unsigned short* __restrict__ Qb, const unsigned short* __restrict__ Kb,
           const unsigned short* __restrict__ Vb, const float* __restrict__ rel_bias,
           const float* __restrict__ attn_mask, unsigned short* __restrict__ Ob) {
    const int HD = 1024;            // H*DK
    const int S  = 1024;
    const int bh = blockIdx.x;
    const int b  = bh >> 4;
    const int h  = bh & 15;
    const int qbase = blockIdx.y * 64;
    const int tid  = threadIdx.x;
    const int wave = tid >> 5;
    const int lane = tid & 31;
    const int lm   = lane & 15;
    const int halfl = lane >> 4;

    __shared__ unsigned short kT[64][32];     // K tile transposed [d][kv], 4 KB
    __shared__ unsigned short vT[32][64];     // V tile row-major, 4 KB
    __shared__ unsigned short pT[4][16][32];  // per-wave P tiles, 4 KB
    __shared__ float biasW[32 * 16];          // rel_bias [NB][H], 2 KB

    for (int i = tid; i < 32 * 16; i += 128) biasW[i] = rel_bias[i];

    // ---- Q fragments (A layout) loaded straight from global ----
    Frag16 qa[2];
    {
        const size_t qoff = (size_t)(b * S + qbase + wave * 16 + lm) * HD + h * 64;
        #pragma unroll
        for (int kb = 0; kb < 2; ++kb)
            #pragma unroll
            for (int v = 0; v < 8; ++v) {
                int kk = kb * 32 + (v >> 2) * 16 + halfl * 8 + (v & 3) * 2;
                qa[kb].u[v] = *(const unsigned int*)(Qb + qoff + kk);
            }
    }

    v8f ctx[4];
    #pragma unroll
    for (int t = 0; t < 4; ++t) ctx[t] = (v8f){0,0,0,0,0,0,0,0};
    float mrow[8], lrow[8];
    #pragma unroll
    for (int r = 0; r < 8; ++r) { mrow[r] = -1e30f; lrow[r] = 0.0f; }

    for (int kv0 = 0; kv0 < S; kv0 += 32) {
        __syncthreads();  // protect LDS tiles from previous iteration readers
        // ---- cooperative load: K tile (transposed into kT), V tile (async) ----
        #pragma unroll
        for (int i = 0; i < 2; ++i) {
            int c  = tid + i * 128;    // 0..255 chunks of 8 bf16
            int r  = c >> 3;           // kv row 0..31
            int d0 = (c & 7) * 8;
            const size_t base = (size_t)(b * S + kv0 + r) * HD + h * 64 + d0;
            uint4 kx = *(const uint4*)(Kb + base);
            const unsigned short* ks = (const unsigned short*)&kx;
            #pragma unroll
            for (int j = 0; j < 8; ++j) kT[d0 + j][r] = ks[j];
            copy16_g2l(Vb + base, &vT[r][d0]);
        }
        wait_g2l();
        __syncthreads();

        // ---- S tile = Q @ K^T  (16 q-rows x 32 kv-cols per wave) ----
        v8f sc[2];
        sc[0] = (v8f){0,0,0,0,0,0,0,0};
        sc[1] = (v8f){0,0,0,0,0,0,0,0};
        #pragma unroll
        for (int u = 0; u < 2; ++u) {
            #pragma unroll
            for (int kb = 0; kb < 2; ++kb) {
                Frag16 kf;
                #pragma unroll
                for (int v = 0; v < 8; ++v) {
                    int kk = kb * 32 + (v >> 2) * 16 + halfl * 8 + (v & 3) * 2;
                    kf.u[v] = (unsigned int)kT[kk][u * 16 + lm] | ((unsigned int)kT[kk + 1][u * 16 + lm] << 16);
                }
                sc[u] = __builtin_amdgcn_wmma_f32_16x16x32_bf16(
                    false, qa[kb].v, false, kf.v, (short)0, sc[u], false, false);
            }
        }

        // masks (per-lane kv column, shared by all rows)
        const float msk0 = (1.0f - attn_mask[b * S + kv0 + lm]) * -10000.0f;
        const float msk1 = (1.0f - attn_mask[b * S + kv0 + 16 + lm]) * -10000.0f;

        // ---- bias + online softmax, row by row (row uniform per half) ----
        #pragma unroll
        for (int r = 0; r < 8; ++r) {
            const int qg = qbase + wave * 16 + r + 8 * halfl;
            const int kv0g = kv0 + lm;
            float s0 = sc[0][r] + biasW[relbucket(qg, kv0g) * 16 + h] + msk0;
            float s1 = sc[1][r] + biasW[relbucket(qg, kv0g + 16) * 16 + h] + msk1;
            float mx = fmaxf(s0, s1);
            mx = fmaxf(mx, SWZF(mx, 8));
            mx = fmaxf(mx, SWZF(mx, 4));
            mx = fmaxf(mx, SWZF(mx, 2));
            mx = fmaxf(mx, SWZF(mx, 1));
            const float mnew  = fmaxf(mrow[r], mx);
            const float alpha = __expf(mrow[r] - mnew);
            mrow[r] = mnew;
            const float p0 = __expf(s0 - mnew);
            const float p1 = __expf(s1 - mnew);
            float rs = p0 + p1;
            rs += SWZF(rs, 8);
            rs += SWZF(rs, 4);
            rs += SWZF(rs, 2);
            rs += SWZF(rs, 1);
            lrow[r] = lrow[r] * alpha + rs;
            #pragma unroll
            for (int t = 0; t < 4; ++t) ctx[t][r] *= alpha;
            pT[wave][r + 8 * halfl][lm]      = f2bf(p0);
            pT[wave][r + 8 * halfl][16 + lm] = f2bf(p1);
        }

        // ---- ctx += P @ V  (per-wave pT, shared vT; same-wave RAW -> dscnt) ----
        Frag16 pa;
        #pragma unroll
        for (int v = 0; v < 8; ++v) {
            int kk = (v >> 2) * 16 + halfl * 8 + (v & 3) * 2;
            pa.u[v] = *(const unsigned int*)&pT[wave][lm][kk];
        }
        #pragma unroll
        for (int t = 0; t < 4; ++t) {
            Frag16 vf;
            #pragma unroll
            for (int v = 0; v < 8; ++v) {
                int kk = (v >> 2) * 16 + halfl * 8 + (v & 3) * 2;
                vf.u[v] = (unsigned int)vT[kk][t * 16 + lm] | ((unsigned int)vT[kk + 1][t * 16 + lm] << 16);
            }
            ctx[t] = __builtin_amdgcn_wmma_f32_16x16x32_bf16(
                false, pa.v, false, vf.v, (short)0, ctx[t], false, false);
        }
    }

    // ---- normalize and store ctx as bf16 in [B*S, H*DK] layout ----
    #pragma unroll
    for (int t = 0; t < 4; ++t) {
        #pragma unroll
        for (int r = 0; r < 8; ++r) {
            const int qg = qbase + wave * 16 + r + 8 * halfl;
            const float o = ctx[t][r] / lrow[r];
            Ob[(size_t)(b * S + qg) * HD + h * 64 + t * 16 + lm] = f2bf(o);
        }
    }
}

// ---------------------------------------------------------------------------
// Orchestration
// ---------------------------------------------------------------------------
extern "C" void kernel_launch(void* const* d_in, const int* in_sizes, int n_in,
                              void* d_out, int out_size, void* d_ws, size_t ws_size,
                              hipStream_t stream) {
    (void)in_sizes; (void)n_in; (void)out_size; (void)ws_size;
    const float* hidden = (const float*)d_in[0];
    const float* amask  = (const float*)d_in[1];
    const float* ln1    = (const float*)d_in[2];
    const float* wq     = (const float*)d_in[3];
    const float* wk     = (const float*)d_in[4];
    const float* wv     = (const float*)d_in[5];
    const float* wo     = (const float*)d_in[6];
    const float* relb   = (const float*)d_in[7];
    const float* ln2    = (const float*)d_in[8];
    const float* wi     = (const float*)d_in[9];
    const float* woff   = (const float*)d_in[10];

    const size_t MB = 1ull << 20;
    char* ws = (char*)d_ws;
    unsigned short* x_b    = (unsigned short*)(ws + 0 * MB);    // 8 MB  [4096,1024]
    unsigned short* q_b    = (unsigned short*)(ws + 8 * MB);    // 8 MB
    unsigned short* k_b    = (unsigned short*)(ws + 16 * MB);   // 8 MB
    unsigned short* v_b    = (unsigned short*)(ws + 24 * MB);   // 8 MB
    unsigned short* ctx_b  = (unsigned short*)(ws + 32 * MB);   // 8 MB
    unsigned short* y_b    = (unsigned short*)(ws + 40 * MB);   // 8 MB
    float*          hid2   = (float*)(ws + 48 * MB);            // 16 MB [4096,1024] f32
    unsigned short* ff_b   = (unsigned short*)(ws + 64 * MB);   // 32 MB [4096,4096]
    unsigned short* wq_b   = (unsigned short*)(ws + 96 * MB);   // 2 MB
    unsigned short* wk_b   = (unsigned short*)(ws + 98 * MB);
    unsigned short* wv_b   = (unsigned short*)(ws + 100 * MB);
    unsigned short* wo_b   = (unsigned short*)(ws + 102 * MB);
    unsigned short* wi_b   = (unsigned short*)(ws + 104 * MB);  // 8 MB
    unsigned short* woff_b = (unsigned short*)(ws + 112 * MB);  // 8 MB

    // weight conversion f32 -> bf16
    cvt_f32_bf16<<<1024, 256, 0, stream>>>(wq,   wq_b,   1024 * 1024);
    cvt_f32_bf16<<<1024, 256, 0, stream>>>(wk,   wk_b,   1024 * 1024);
    cvt_f32_bf16<<<1024, 256, 0, stream>>>(wv,   wv_b,   1024 * 1024);
    cvt_f32_bf16<<<1024, 256, 0, stream>>>(wo,   wo_b,   1024 * 1024);
    cvt_f32_bf16<<<2048, 256, 0, stream>>>(wi,   wi_b,   1024 * 4096);
    cvt_f32_bf16<<<2048, 256, 0, stream>>>(woff, woff_b, 4096 * 1024);

    // x = RMSNorm(hidden) * ln1_w   (bf16)
    rmsnorm_bf16<<<4096, 256, 0, stream>>>(hidden, ln1, x_b, 1024);

    // Q, K, V projections
    gemm_bf16<0><<<dim3(32, 8), 256, 0, stream>>>(x_b, wq_b, q_b, nullptr, 4096, 1024, 1024);
    gemm_bf16<0><<<dim3(32, 8), 256, 0, stream>>>(x_b, wk_b, k_b, nullptr, 4096, 1024, 1024);
    gemm_bf16<0><<<dim3(32, 8), 256, 0, stream>>>(x_b, wv_b, v_b, nullptr, 4096, 1024, 1024);

    // flash attention with relative-position bias + mask
    flash_attn<<<dim3(64, 16), 128, 0, stream>>>(q_b, k_b, v_b, relb, amask, ctx_b);

    // attn_out @ wo + hidden residual -> hid2 (f32)
    gemm_bf16<2><<<dim3(32, 8), 256, 0, stream>>>(ctx_b, wo_b, hid2, hidden, 4096, 1024, 1024);

    // y = RMSNorm(hid2) * ln2_w
    rmsnorm_bf16<<<4096, 256, 0, stream>>>(hid2, ln2, y_b, 1024);

    // FFN: relu(y @ wi) @ wo_ff + hid2 -> d_out (f32)
    gemm_bf16<1><<<dim3(32, 32), 256, 0, stream>>>(y_b, wi_b, ff_b, nullptr, 4096, 4096, 1024);
    gemm_bf16<2><<<dim3(32, 8), 256, 0, stream>>>(ff_b, woff_b, (float*)d_out, hid2, 4096, 1024, 4096);
}
